// Layer_87359634801306
// MI455X (gfx1250) — matrix operand order repaired and checked by reading, not compiled, not run on hardware
//
#include <hip/hip_runtime.h>

// ---------------------------------------------------------------------------
// Types for CDNA5 WMMA (wave32)
// ---------------------------------------------------------------------------
typedef __attribute__((ext_vector_type(16))) __bf16 v16bf;
typedef __attribute__((ext_vector_type(8)))  float  v8f;
typedef __attribute__((ext_vector_type(4)))  unsigned int u32x4;

union FragU { v16bf v; u32x4 u[2]; };

// A fragment (16x32 bf16): lane lr holds row (row0+lr); K chunks
// [k0 + hi*8, +8) and [k0 + 16 + hi*8, +8)   (ISA 7.12.2, 16-bit A 16x32)
__device__ __forceinline__ v16bf load_frag_a(const __bf16* base, int ld,
                                             int row0, int k0, int lr, int hi) {
  const __bf16* p = base + (size_t)(row0 + lr) * ld + k0 + hi * 8;
  FragU f;
  f.u[0] = *(const u32x4*)p;
  f.u[1] = *(const u32x4*)(p + 16);
  return f.v;
}

// B fragment (32x16 bf16) from row-major B^T [N][K]: lane lr holds column
// (col0+lr); K chunks [k0 + hi*8, +8) and [k0 + 16 + hi*8, +8)
__device__ __forceinline__ v16bf load_frag_b(const __bf16* base, int ld,
                                             int col0, int k0, int lr, int hi) {
  const __bf16* p = base + (size_t)(col0 + lr) * ld + k0 + hi * 8;
  FragU f;
  f.u[0] = *(const u32x4*)p;
  f.u[1] = *(const u32x4*)(p + 16);
  return f.v;
}

__device__ __forceinline__ v8f wmma_bf16(v16bf a, v16bf b, v8f c) {
  return __builtin_amdgcn_wmma_f32_16x16x32_bf16(
      /*neg_a=*/false, a, /*neg_b=*/false, b,
      /*c_mod=*/(short)0, c, /*reuse_a=*/false, /*reuse_b=*/false);
}

// Async global->LDS copy, 16 bytes per lane (ASYNCcnt-tracked, no VGPR data)
__device__ __forceinline__ void async_copy_b128(unsigned lds_addr,
                                                const __bf16* gaddr) {
  asm volatile("global_load_async_to_lds_b128 %0, %1, off"
               :: "v"(lds_addr), "v"(gaddr) : "memory");
}

// ---------------------------------------------------------------------------
// Problem constants
// ---------------------------------------------------------------------------
#define BB 16
#define TT 512
#define DD 1024
#define HH 16
#define HD 64
#define MROWS (BB * TT)   // 8192

// ---------------------------------------------------------------------------
// Weight convert / transpose:  W[K][N] f32  ->  Wt[N][K] bf16
// ---------------------------------------------------------------------------
__global__ void k_transpose_bf16(const float* __restrict__ W,
                                 __bf16* __restrict__ Wt, int K, int N) {
  int i = blockIdx.x * 256 + threadIdx.x;        // over N*K, contiguous in K
  int n = i / K;
  int k = i - n * K;
  Wt[i] = (__bf16)W[(size_t)k * N + n];
}

// Wq/Wk/Wv [H][D][HD] f32 -> Wt[(h*HD+e)][D] bf16
__global__ void k_qkv_transpose_bf16(const float* __restrict__ W,
                                     __bf16* __restrict__ Wt) {
  int i = blockIdx.x * 256 + threadIdx.x;        // over 1024*1024
  int n = i >> 10;                               // h*HD+e
  int d = i & 1023;
  int h = n >> 6;
  int e = n & 63;
  Wt[i] = (__bf16)W[((size_t)h * DD + d) * HD + e];
}

// v [B,T,H,HD] bf16 -> vT [B,H,HD,T] bf16
__global__ void k_permute_v(const __bf16* __restrict__ V,
                            __bf16* __restrict__ Vt) {
  int i = blockIdx.x * 256 + threadIdx.x;        // over B*H*HD*T = 8388608
  int s  = i & (TT - 1);
  int e  = (i >> 9) & (HD - 1);
  int bh = i >> 15;
  int b  = bh >> 4;
  int h  = bh & 15;
  Vt[i] = V[(((size_t)b * TT + s) * HH + h) * HD + e];
}

// ---------------------------------------------------------------------------
// LayerNorm over D=1024, write bf16.  One block (256 thr) per row.
// ---------------------------------------------------------------------------
__global__ __launch_bounds__(256) void k_layernorm_bf16(
    const float* __restrict__ x, const float* __restrict__ g,
    const float* __restrict__ bta, __bf16* __restrict__ out) {
  int row = blockIdx.x;
  int tid = threadIdx.x;
  const float* xr = x + (size_t)row * DD;
  float v[4], s = 0.f, s2 = 0.f;
#pragma unroll
  for (int i = 0; i < 4; i++) {
    v[i] = xr[tid + i * 256];
    s += v[i];
    s2 += v[i] * v[i];
  }
#pragma unroll
  for (int m = 1; m < 32; m <<= 1) {
    s  += __shfl_xor(s,  m, 32);
    s2 += __shfl_xor(s2, m, 32);
  }
  __shared__ float rs[8], rs2[8];
  int w = tid >> 5;
  if ((tid & 31) == 0) { rs[w] = s; rs2[w] = s2; }
  __syncthreads();
  float ts = 0.f, ts2 = 0.f;
#pragma unroll
  for (int i = 0; i < 8; i++) { ts += rs[i]; ts2 += rs2[i]; }
  float mean = ts * (1.0f / DD);
  float var  = ts2 * (1.0f / DD) - mean * mean;
  float rstd = rsqrtf(var + 1e-5f);
#pragma unroll
  for (int i = 0; i < 4; i++) {
    int col = tid + i * 256;
    out[(size_t)row * DD + col] =
        (__bf16)((v[i] - mean) * rstd * g[col] + bta[col]);
  }
}

// ---------------------------------------------------------------------------
// TN GEMM:  C[M,N] = A[M,K](bf16) * Bt[N,K]^T(bf16) + bias (+res) (relu)
// Workgroup 256 thr (8 waves) -> 128x128 tile; wave -> 32x64 (2x4 WMMA tiles).
// K is consumed in 64-wide slabs, double-buffered in LDS and filled with
// ASYNCcnt-tracked global_load_async_to_lds_b128 (CDNA5 async copy path).
// ---------------------------------------------------------------------------
template <bool BIAS, bool RELU, bool RES, bool OUTF, bool OUTB>
__global__ __launch_bounds__(256) void k_gemm_tn(
    const __bf16* __restrict__ A, const __bf16* __restrict__ Bt,
    const float* __restrict__ bias, const float* __restrict__ res,
    float* __restrict__ Cf, __bf16* __restrict__ Cb, int M, int N, int K) {
  __shared__ __bf16 shA[2][128][64];   // 2 x 16 KB
  __shared__ __bf16 shB[2][128][64];   // 2 x 16 KB

  const int tid  = threadIdx.x;
  const int lane = tid & 31;
  const int w    = tid >> 5;
  const int lr   = lane & 15;
  const int hi   = lane >> 4;
  const int wm   = w & 3;
  const int wn   = w >> 2;
  const int mblk = blockIdx.y * 128;
  const int nblk = blockIdx.x * 128;

  // staging assignment: 2 threads per row, each copies 32 bf16 (4 x B128)
  const int srow = tid >> 1;           // 0..127
  const int sseg = (tid & 1) * 32;     // 0 or 32
  const __bf16* gA = A  + (size_t)(mblk + srow) * K + sseg;
  const __bf16* gB = Bt + (size_t)(nblk + srow) * K + sseg;
  // flat->LDS: low 32 bits of a generic pointer to LDS are the byte offset
  const unsigned ldsA0 = (unsigned)(uintptr_t)&shA[0][srow][sseg];
  const unsigned ldsB0 = (unsigned)(uintptr_t)&shB[0][srow][sseg];
  const unsigned bufBytes = 128u * 64u * 2u;   // 16 KB per buffer

  auto stage = [&](int ks, int buf) {
    const __bf16* pa = gA + (size_t)ks * 64;
    const __bf16* pb = gB + (size_t)ks * 64;
    const unsigned la = ldsA0 + (unsigned)buf * bufBytes;
    const unsigned lb = ldsB0 + (unsigned)buf * bufBytes;
#pragma unroll
    for (int c = 0; c < 4; c++) {
      async_copy_b128(la + c * 16u, pa + c * 8);
      async_copy_b128(lb + c * 16u, pb + c * 8);
    }
  };

  v8f acc[2][4] = {};
  const int NK = K >> 6;               // 64-wide slabs

  stage(0, 0);
  for (int ks = 0; ks < NK; ks++) {
    const int buf = ks & 1;
    if (ks + 1 < NK) {
      stage(ks + 1, buf ^ 1);
      // older slab's 8 copies complete first (async loads retire in order)
      asm volatile("s_wait_asynccnt 8" ::: "memory");
    } else {
      asm volatile("s_wait_asynccnt 0" ::: "memory");
    }
    __syncthreads();

    const __bf16* sa = &shA[buf][0][0];
    const __bf16* sb = &shB[buf][0][0];
#pragma unroll
    for (int kk = 0; kk < 64; kk += 32) {
      v16bf a0  = load_frag_a(sa, 64, wm * 32,      kk, lr, hi);
      v16bf a1  = load_frag_a(sa, 64, wm * 32 + 16, kk, lr, hi);
      v16bf bb0 = load_frag_b(sb, 64, wn * 64,      kk, lr, hi);
      v16bf bb1 = load_frag_b(sb, 64, wn * 64 + 16, kk, lr, hi);
      v16bf bb2 = load_frag_b(sb, 64, wn * 64 + 32, kk, lr, hi);
      v16bf bb3 = load_frag_b(sb, 64, wn * 64 + 48, kk, lr, hi);
      acc[0][0] = wmma_bf16(a0, bb0, acc[0][0]);
      acc[0][1] = wmma_bf16(a0, bb1, acc[0][1]);
      acc[0][2] = wmma_bf16(a0, bb2, acc[0][2]);
      acc[0][3] = wmma_bf16(a0, bb3, acc[0][3]);
      acc[1][0] = wmma_bf16(a1, bb0, acc[1][0]);
      acc[1][1] = wmma_bf16(a1, bb1, acc[1][1]);
      acc[1][2] = wmma_bf16(a1, bb2, acc[1][2]);
      acc[1][3] = wmma_bf16(a1, bb3, acc[1][3]);
    }
    __syncthreads();   // protect buffer before it is refilled
  }

  const int m0 = mblk + wm * 32;
  const int n0 = nblk + wn * 64;
#pragma unroll
  for (int i = 0; i < 2; i++) {
#pragma unroll
    for (int j = 0; j < 4; j++) {
      const int n = n0 + j * 16 + lr;
      const float bvv = BIAS ? bias[n] : 0.0f;
#pragma unroll
      for (int r = 0; r < 8; r++) {
        const int m = m0 + i * 16 + r + hi * 8;
        const size_t idx = (size_t)m * N + n;
        float c = acc[i][j][r] + bvv;
        if (RES)  c += res[idx];
        if (RELU) c = fmaxf(c, 0.0f);
        if (OUTF) Cf[idx] = c;
        if (OUTB) Cb[idx] = (__bf16)c;
      }
    }
  }
}

// ---------------------------------------------------------------------------
// Flash attention (reference semantics: wei[t,s] = k[t].q[s], mask s<=t,
// softmax over s, o = wei @ v).  One workgroup = 8 waves = 128 t-rows of one
// (b,h).  K/Q/O in [B,T,H*HD] bf16, Vt in [B,H,HD,T] bf16.
// ---------------------------------------------------------------------------
__global__ __launch_bounds__(256) void k_attention(
    const __bf16* __restrict__ Kb, const __bf16* __restrict__ Qb,
    const __bf16* __restrict__ Vt, __bf16* __restrict__ O) {
  __shared__ __bf16 lds[8][16][32];   // wave-private 16x32 P tile (8 KB)

  const int lane = threadIdx.x & 31;
  const int w    = threadIdx.x >> 5;
  const int lr   = lane & 15;
  const int hi   = lane >> 4;
  const int bh   = blockIdx.x >> 2;
  const int tb   = blockIdx.x & 3;
  const int b    = bh >> 4;
  const int h    = bh & 15;
  const int t0   = tb * 128 + w * 16;

  const size_t rowbase = ((size_t)b * TT) * DD + (size_t)h * HD;
  const size_t vtbase  = (((size_t)b * HH + h) * HD) * TT;

  // K rows t0..t0+15 (A fragments), resident across the s loop
  const v16bf ak0 = load_frag_a(Kb + rowbase, DD, t0, 0,  lr, hi);
  const v16bf ak1 = load_frag_a(Kb + rowbase, DD, t0, 32, lr, hi);

  v8f accO[4] = {};
  float mrow[8], lrow[8];
#pragma unroll
  for (int r = 0; r < 8; r++) { mrow[r] = -3.0e38f; lrow[r] = 0.0f; }

  const v8f zero = {};
  for (int s0 = 0; s0 <= t0 + 15; s0 += 32) {
    // scores for 32 s-columns: two 16x16 tiles, each K=64 (two WMMAs)
    v16bf q00 = load_frag_b(Qb + rowbase, DD, s0,      0,  lr, hi);
    v16bf q01 = load_frag_b(Qb + rowbase, DD, s0,      32, lr, hi);
    v16bf q10 = load_frag_b(Qb + rowbase, DD, s0 + 16, 0,  lr, hi);
    v16bf q11 = load_frag_b(Qb + rowbase, DD, s0 + 16, 32, lr, hi);
    v8f S0 = wmma_bf16(ak1, q01, wmma_bf16(ak0, q00, zero));
    v8f S1 = wmma_bf16(ak1, q11, wmma_bf16(ak0, q10, zero));

    float scl[8];
#pragma unroll
    for (int r = 0; r < 8; r++) {
      const int trow = t0 + r + hi * 8;
      float v0 = (s0 + lr      <= trow) ? S0[r] : -3.0e38f;
      float v1 = (s0 + 16 + lr <= trow) ? S1[r] : -3.0e38f;
      // row max across the 16 lanes of this half-wave
      float mx = fmaxf(v0, v1);
      mx = fmaxf(mx, __shfl_xor(mx, 1, 32));
      mx = fmaxf(mx, __shfl_xor(mx, 2, 32));
      mx = fmaxf(mx, __shfl_xor(mx, 4, 32));
      mx = fmaxf(mx, __shfl_xor(mx, 8, 32));
      const float nm = fmaxf(mrow[r], mx);
      const float p0 = __expf(v0 - nm);
      const float p1 = __expf(v1 - nm);
      float rsum = p0 + p1;
      rsum += __shfl_xor(rsum, 1, 32);
      rsum += __shfl_xor(rsum, 2, 32);
      rsum += __shfl_xor(rsum, 4, 32);
      rsum += __shfl_xor(rsum, 8, 32);
      const float sc = __expf(mrow[r] - nm);
      lrow[r] = lrow[r] * sc + rsum;
      mrow[r] = nm;
      scl[r]  = sc;
      // stash P in C-layout into the wave-private LDS tile (bf16)
      lds[w][r + hi * 8][lr]      = (__bf16)p0;
      lds[w][r + hi * 8][16 + lr] = (__bf16)p1;
    }
#pragma unroll
    for (int j = 0; j < 4; j++)
#pragma unroll
      for (int r = 0; r < 8; r++) accO[j][r] *= scl[r];

    // wave-local LDS fence: C-layout stores -> A-layout reloads below
    asm volatile("s_wait_dscnt 0" ::: "memory");

    FragU pf;
    pf.u[0] = *(const u32x4*)&lds[w][lr][hi * 8];
    pf.u[1] = *(const u32x4*)&lds[w][lr][16 + hi * 8];

    // o += P @ V   (B fragments from Vt rows = e-columns, K = s contiguous)
#pragma unroll
    for (int j = 0; j < 4; j++) {
      v16bf bv = load_frag_b(Vt + vtbase, TT, j * 16, s0, lr, hi);
      accO[j] = wmma_bf16(pf.v, bv, accO[j]);
    }
  }

#pragma unroll
  for (int j = 0; j < 4; j++) {
#pragma unroll
    for (int r = 0; r < 8; r++) {
      const float o = accO[j][r] / lrow[r];
      O[rowbase + (size_t)(t0 + r + hi * 8) * DD + j * 16 + lr] = (__bf16)o;
    }
  }
}

// ---------------------------------------------------------------------------
// Host: orchestrate the whole transformer block on `stream`
// ---------------------------------------------------------------------------
extern "C" void kernel_launch(void* const* d_in, const int* in_sizes, int n_in,
                              void* d_out, int out_size, void* d_ws,
                              size_t ws_size, hipStream_t stream) {
  (void)in_sizes; (void)n_in; (void)out_size; (void)ws_size;
  const float* x    = (const float*)d_in[0];
  const float* ga   = (const float*)d_in[1];
  const float* ba   = (const float*)d_in[2];
  const float* Win  = (const float*)d_in[3];
  const float* bin_ = (const float*)d_in[4];
  const float* Wk   = (const float*)d_in[5];
  const float* bk   = (const float*)d_in[6];
  const float* Wq   = (const float*)d_in[7];
  const float* bq   = (const float*)d_in[8];
  const float* Wv   = (const float*)d_in[9];
  const float* bv   = (const float*)d_in[10];
  const float* Wout = (const float*)d_in[11];
  const float* bout = (const float*)d_in[12];
  const float* gf   = (const float*)d_in[13];
  const float* bf_  = (const float*)d_in[14];
  const float* W1   = (const float*)d_in[15];
  const float* b1   = (const float*)d_in[16];
  const float* W2   = (const float*)d_in[17];
  const float* b2   = (const float*)d_in[18];

  char* ws = (char*)d_ws;
  size_t off = 0;
  auto take = [&](size_t bytes) -> char* {
    char* p = ws + off;
    off += (bytes + 255) & ~(size_t)255;
    return p;
  };

  const size_t actB = (size_t)MROWS * DD * sizeof(__bf16);       // 16 MB
  __bf16* WinT  = (__bf16*)take((size_t)DD * DD * 2);
  __bf16* WqT   = (__bf16*)take((size_t)DD * DD * 2);
  __bf16* WkT   = (__bf16*)take((size_t)DD * DD * 2);
  __bf16* WvT   = (__bf16*)take((size_t)DD * DD * 2);
  __bf16* WoT   = (__bf16*)take((size_t)DD * DD * 2);
  __bf16* W1T   = (__bf16*)take((size_t)4 * DD * DD * 2);
  __bf16* W2T   = (__bf16*)take((size_t)4 * DD * DD * 2);
  __bf16* a_ln  = (__bf16*)take(actB);            // reused for ln2
  __bf16* a_h   = (__bf16*)take(actB);            // reused for attention out
  __bf16* a_q   = (__bf16*)take(actB);
  __bf16* a_k   = (__bf16*)take(actB);
  __bf16* a_v   = (__bf16*)take(actB);
  __bf16* a_vT  = (__bf16*)take(actB);
  float*  x2    = (float*)take((size_t)MROWS * DD * sizeof(float));
  __bf16* a_f1  = (__bf16*)take((size_t)MROWS * 4 * DD * sizeof(__bf16));
  __bf16* a_ln2 = a_ln;
  __bf16* a_o   = a_h;

  // --- weight convert/transpose (bf16) ---
  k_transpose_bf16<<<(DD * DD) / 256, 256, 0, stream>>>(Win, WinT, DD, DD);
  k_qkv_transpose_bf16<<<(DD * DD) / 256, 256, 0, stream>>>(Wq, WqT);
  k_qkv_transpose_bf16<<<(DD * DD) / 256, 256, 0, stream>>>(Wk, WkT);
  k_qkv_transpose_bf16<<<(DD * DD) / 256, 256, 0, stream>>>(Wv, WvT);
  k_transpose_bf16<<<(DD * DD) / 256, 256, 0, stream>>>(Wout, WoT, DD, DD);
  k_transpose_bf16<<<(4 * DD * DD) / 256, 256, 0, stream>>>(W1, W1T, DD, 4 * DD);
  k_transpose_bf16<<<(4 * DD * DD) / 256, 256, 0, stream>>>(W2, W2T, 4 * DD, DD);

  // --- attention sub-block ---
  k_layernorm_bf16<<<MROWS, 256, 0, stream>>>(x, ga, ba, a_ln);

  dim3 g1(DD / 128, MROWS / 128);   // N=1024 GEMMs
  k_gemm_tn<true, false, false, false, true><<<g1, 256, 0, stream>>>(
      a_ln, WinT, bin_, nullptr, nullptr, a_h, MROWS, DD, DD);
  k_gemm_tn<true, false, false, false, true><<<g1, 256, 0, stream>>>(
      a_h, WqT, bq, nullptr, nullptr, a_q, MROWS, DD, DD);
  k_gemm_tn<true, false, false, false, true><<<g1, 256, 0, stream>>>(
      a_h, WkT, bk, nullptr, nullptr, a_k, MROWS, DD, DD);
  k_gemm_tn<true, false, false, false, true><<<g1, 256, 0, stream>>>(
      a_h, WvT, bv, nullptr, nullptr, a_v, MROWS, DD, DD);

  k_permute_v<<<(BB * HH * HD * TT) / 256, 256, 0, stream>>>(a_v, a_vT);

  k_attention<<<BB * HH * (TT / 128), 256, 0, stream>>>(a_k, a_q, a_vT, a_o);

  k_gemm_tn<true, false, true, true, false><<<g1, 256, 0, stream>>>(
      a_o, WoT, bout, x, x2, nullptr, MROWS, DD, DD);

  // --- FFN sub-block ---
  k_layernorm_bf16<<<MROWS, 256, 0, stream>>>(x2, gf, bf_, a_ln2);

  dim3 g2(4 * DD / 128, MROWS / 128);
  k_gemm_tn<true, true, false, false, true><<<g2, 256, 0, stream>>>(
      a_ln2, W1T, b1, nullptr, nullptr, a_f1, MROWS, 4 * DD, DD);

  k_gemm_tn<true, false, true, true, false><<<g1, 256, 0, stream>>>(
      a_f1, W2T, b2, x2, (float*)d_out, nullptr, MROWS, DD, 4 * DD);
}
